// BPNN_2164663517680
// MI455X (gfx1250) — matrix operand order repaired and checked by reading, not compiled, not run on hardware
//
#include <hip/hip_runtime.h>
#include <hip/hip_bf16.h>

typedef __attribute__((ext_vector_type(16))) _Float16 v16h;
typedef __attribute__((ext_vector_type(8)))  float    v8f;

#define LDSTR 136            // padded row stride (halves): 128 data + 8 pad
#define HDIM  128
#define IN_DIM 120
// per-element packed-weight stride (halves): 6 full 128x128 mats + [128x16] w4 image
#define PER_E (6 * HDIM * HDIM + HDIM * 16)

// Hardware tanh (V_TANH_F32 on gfx1250) with portable fallback.
__device__ __forceinline__ float htanh(float x) {
#if __has_builtin(__builtin_amdgcn_tanhf)
    return __builtin_amdgcn_tanhf(x);
#else
    return tanhf(x);
#endif
}

// ---------------------------------------------------------------------------
// Fragment loaders
// ---------------------------------------------------------------------------
// A-matrix 16x32 f16 (ISA 7.12.2): lane m=l%16, kh=l/16; VGPR0..3 hold
// K = kh*8 + {0..7}, VGPR4..7 hold K = 16 + kh*8 + {0..7}.  With row-major
// f16 activations in LDS this is two 16B loads per lane.
__device__ __forceinline__ v16h load_afrag(const _Float16* Abase, int lane, int kt) {
    int m  = lane & 15;
    int kh = lane >> 4;
    const _Float16* row = Abase + m * LDSTR;
    int c0 = kt * 32 + kh * 8;
    union { uint4 q[2]; v16h h; } u;
    u.q[0] = *(const uint4*)(row + c0);
    u.q[1] = *(const uint4*)(row + c0 + 16);
    return u.h;
}

// B-matrix 32x16 f16: lane n=l%16, kh=l/16; halves 0..7 hold K = kh*8 + {0..7},
// halves 8..15 hold K = 16 + kh*8 + {0..7} (same as A), gathered from a
// row-major f16 [128 x 128] weight image.
__device__ __forceinline__ v16h load_bfrag(const _Float16* Bmat, int lane, int kt, int nt) {
    const _Float16* p = Bmat + nt * 16 + (lane & 15);
    int kb = kt * 32 + (lane >> 4) * 8;
    v16h h;
#pragma unroll
    for (int s = 0; s < 16; ++s) {
        int k = kb + (s < 8 ? s : 8 + s);
        h[s] = p[(size_t)k * HDIM];
    }
    return h;
}

// Same, but for a narrow [128 x 16] B image (single N-tile, e.g. W4 GEMV).
__device__ __forceinline__ v16h load_bfrag_n16(const _Float16* Bmat, int lane, int kt) {
    const _Float16* p = Bmat + (lane & 15);
    int kb = kt * 32 + (lane >> 4) * 8;
    v16h h;
#pragma unroll
    for (int s = 0; s < 16; ++s) {
        int k = kb + (s < 8 ? s : 8 + s);
        h[s] = p[(size_t)k * 16];
    }
    return h;
}

// One 16x16 output tile: accumulate over K = 128 (4 steps of 32).
__device__ __forceinline__ v8f tile_gemm(const _Float16* Abase, const _Float16* Bmat,
                                         int lane, int nt, v8f c) {
#pragma unroll
    for (int kt = 0; kt < 4; ++kt) {
        v16h a = load_afrag(Abase, lane, kt);
        v16h b = load_bfrag(Bmat, lane, kt, nt);
        c = __builtin_amdgcn_wmma_f32_16x16x32_f16(false, a, false, b,
                                                   (short)0, c, false, false);
    }
    return c;
}

// Forward layer: H = tanh(A @ W + b), all 8 N-tiles, result to LDS f16.
__device__ __forceinline__ void fwd_layer(const _Float16* Aw, _Float16* Hw,
                                          const _Float16* Wp, const float* bvec, int lane) {
#pragma unroll 1
    for (int nt = 0; nt < 8; ++nt) {
        v8f acc = {0.f, 0.f, 0.f, 0.f, 0.f, 0.f, 0.f, 0.f};
        acc = tile_gemm(Aw, Wp, lane, nt, acc);
        int n = nt * 16 + (lane & 15);
        float b = bvec[n];
#pragma unroll
        for (int r = 0; r < 8; ++r) {
            int row = (lane >> 4) * 8 + r;
            Hw[row * LDSTR + n] = (_Float16)htanh(acc[r] + b);
        }
    }
}

// Backward layer: Hw holds h (activations); overwritten with
// d = (D @ W^T) * (1 - h^2).
__device__ __forceinline__ void bwd_layer(const _Float16* Dw, _Float16* Hw,
                                          const _Float16* Wtp, int lane) {
#pragma unroll 1
    for (int nt = 0; nt < 8; ++nt) {
        v8f acc = {0.f, 0.f, 0.f, 0.f, 0.f, 0.f, 0.f, 0.f};
        acc = tile_gemm(Dw, Wtp, lane, nt, acc);
        int n = nt * 16 + (lane & 15);
#pragma unroll
        for (int r = 0; r < 8; ++r) {
            int row = (lane >> 4) * 8 + r;
            float h = (float)Hw[row * LDSTR + n];
            Hw[row * LDSTR + n] = (_Float16)(acc[r] * (1.f - h * h));
        }
    }
}

// ---------------------------------------------------------------------------
// Weight pack: f32 -> f16, K padded to 128, plus transposed copies for bwd,
// plus a [128 x 16] image of W4 (column 0 = w4, rest zero) for the WMMA GEMV.
// mat: 0..2 = W1p,W2p,W3p (K x N), 3..5 = transposed+padded, 6 = w4 image.
// ---------------------------------------------------------------------------
__global__ void pack_weights(const float* __restrict__ W1,
                             const float* __restrict__ W2,
                             const float* __restrict__ W3,
                             const float* __restrict__ W4,
                             _Float16* __restrict__ dst) {
    int e   = blockIdx.y;
    int mat = blockIdx.z;
    int idx = blockIdx.x * 256 + threadIdx.x;     // 0..16383
    if (mat == 6) {
        if (idx < HDIM * 16) {
            int k = idx >> 4, n = idx & 15;
            float v = (n == 0) ? W4[e * HDIM + k] : 0.f;
            dst[(size_t)e * PER_E + 6 * HDIM * HDIM + idx] = (_Float16)v;
        }
        return;
    }
    int k = idx >> 7, n = idx & 127;
    float v = 0.f;
    switch (mat) {
        case 0: if (k < IN_DIM) v = W1[(size_t)(e * IN_DIM + k) * HDIM + n]; break;
        case 1: v = W2[(size_t)(e * HDIM + k) * HDIM + n]; break;
        case 2: v = W3[(size_t)(e * HDIM + k) * HDIM + n]; break;
        case 3: if (n < IN_DIM) v = W1[(size_t)(e * IN_DIM + n) * HDIM + k]; break;
        case 4: v = W2[(size_t)(e * HDIM + n) * HDIM + k]; break;
        case 5: v = W3[(size_t)(e * HDIM + n) * HDIM + k]; break;
    }
    dst[(size_t)e * PER_E + (size_t)mat * (HDIM * HDIM) + idx] = (_Float16)v;
}

// ---------------------------------------------------------------------------
// Fused per-atom MLP forward + backward.  Block = 128 atoms = 8 wave32s,
// each wave owns one 16-atom M-tile end-to-end (no inter-wave LDS sharing
// after staging -> one barrier total).
// ---------------------------------------------------------------------------
__global__ void __launch_bounds__(256)
bpnn_kernel(const int* __restrict__ an, const float* __restrict__ fp,
            const int* __restrict__ img, const _Float16* __restrict__ Wpack,
            const float* __restrict__ b1, const float* __restrict__ b2,
            const float* __restrict__ b3, const float* __restrict__ W4,
            const float* __restrict__ b4, float* __restrict__ g,
            float* __restrict__ energy, int n_atoms) {
    extern __shared__ _Float16 smem[];
    _Float16* S  = smem;                       // fingerprints (f16, K-padded)
    _Float16* H1 = smem + 1 * 128 * LDSTR;
    _Float16* H2 = smem + 2 * 128 * LDSTR;
    _Float16* H3 = smem + 3 * 128 * LDSTR;
    int* s_an = (int*)(smem + 4 * 128 * LDSTR);

    int tid  = threadIdx.x;
    int wave = tid >> 5;
    int lane = tid & 31;
    int base = blockIdx.x * 128;

    // Stage fingerprints (zero-pad K 120..127 and out-of-range atoms).
    for (int idx = tid; idx < 128 * 128; idx += 256) {
        int r = idx >> 7, c = idx & 127;
        int atom = base + r;
        float v = (atom < n_atoms && c < IN_DIM) ? fp[(size_t)atom * IN_DIM + c] : 0.f;
        S[r * LDSTR + c] = (_Float16)v;
    }
    if (tid < 128) s_an[tid] = (base + tid < n_atoms) ? an[base + tid] : -1;
    __syncthreads();

    const _Float16* Sw = S  + wave * 16 * LDSTR;
    _Float16* H1w = H1 + wave * 16 * LDSTR;
    _Float16* H2w = H2 + wave * 16 * LDSTR;
    _Float16* H3w = H3 + wave * 16 * LDSTR;

    v8f gacc[8];
#pragma unroll
    for (int nt = 0; nt < 8; ++nt)
        gacc[nt] = (v8f){0.f, 0.f, 0.f, 0.f, 0.f, 0.f, 0.f, 0.f};
    float o8[8];                       // masked per-atom energies (lanes 0,16)
#pragma unroll
    for (int r = 0; r < 8; ++r) o8[r] = 0.f;

#pragma unroll 1
    for (int e = 0; e < 3; ++e) {
        const int elemZ = (e == 0) ? 1 : ((e == 1) ? 6 : 8);
        const _Float16* Wbase = Wpack + (size_t)e * PER_E;
        const _Float16* W1p   = Wbase + 0 * HDIM * HDIM;
        const _Float16* W2p   = Wbase + 1 * HDIM * HDIM;
        const _Float16* W3p   = Wbase + 2 * HDIM * HDIM;
        const _Float16* Wt1p  = Wbase + 3 * HDIM * HDIM;
        const _Float16* Wt2p  = Wbase + 4 * HDIM * HDIM;
        const _Float16* Wt3p  = Wbase + 5 * HDIM * HDIM;
        const _Float16* W4img = Wbase + 6 * HDIM * HDIM;
        __builtin_prefetch(W1p, 0, 2);
        __builtin_prefetch(W2p, 0, 2);
        __builtin_prefetch(W3p, 0, 2);
        const float* b1e = b1 + e * HDIM;
        const float* b2e = b2 + e * HDIM;
        const float* b3e = b3 + e * HDIM;
        const float* w4e = W4 + e * HDIM;
        float b4e = b4[e];

        // Forward
        fwd_layer(Sw,  H1w, W1p, b1e, lane);
        fwd_layer(H1w, H2w, W2p, b2e, lane);
        fwd_layer(H2w, H3w, W3p, b3e, lane);

        // Output layer as a WMMA GEMV against the [128x16] w4 image
        // (only column n==0 valid; it lives in lanes 0 and 16).
        {
            v8f oa = {0.f, 0.f, 0.f, 0.f, 0.f, 0.f, 0.f, 0.f};
#pragma unroll
            for (int kt = 0; kt < 4; ++kt) {
                v16h a = load_afrag(H3w, lane, kt);
                v16h b = load_bfrag_n16(W4img, lane, kt);
                oa = __builtin_amdgcn_wmma_f32_16x16x32_f16(false, a, false, b,
                                                            (short)0, oa, false, false);
            }
            if ((lane & 15) == 0) {
#pragma unroll
                for (int r = 0; r < 8; ++r) {
                    int row = (lane >> 4) * 8 + r;
                    float mv = (s_an[wave * 16 + row] == elemZ) ? 1.f : 0.f;
                    o8[r] += mv * (oa[r] + b4e);
                }
            }
        }

        // d3 = mask * w4 * (1 - h3^2), written in place over H3.
#pragma unroll 1
        for (int nt = 0; nt < 8; ++nt) {
            int n = nt * 16 + (lane & 15);
            float w4n = w4e[n];
#pragma unroll
            for (int r = 0; r < 8; ++r) {
                int row = (lane >> 4) * 8 + r;
                float mv = (s_an[wave * 16 + row] == elemZ) ? 1.f : 0.f;
                float h = (float)H3w[row * LDSTR + n];
                H3w[row * LDSTR + n] = (_Float16)(mv * w4n * (1.f - h * h));
            }
        }

        // Backward GEMMs: d2 <- H2 (in place), d1 <- H1 (in place).
        bwd_layer(H3w, H2w, Wt3p, lane);
        bwd_layer(H2w, H1w, Wt2p, lane);

        // g += d1 @ W1^T  (accumulate across elements in registers)
#pragma unroll
        for (int nt = 0; nt < 8; ++nt)
            gacc[nt] = tile_gemm(H1w, Wt1p, lane, nt, gacc[nt]);
    }

    // Write dE/dfp (only the 120 valid columns, matching g_flat layout).
#pragma unroll
    for (int nt = 0; nt < 8; ++nt) {
        int n = nt * 16 + (lane & 15);
        if (n < IN_DIM) {
#pragma unroll
            for (int r = 0; r < 8; ++r) {
                int row = (lane >> 4) * 8 + r;
                int atom = base + wave * 16 + row;
                if (atom < n_atoms) g[(size_t)atom * IN_DIM + n] = gacc[nt][r];
            }
        }
    }

    // Per-molecule energy reduction (lanes 0 and 16 hold 8 atoms each).
    if ((lane & 15) == 0) {
#pragma unroll
        for (int r = 0; r < 8; ++r) {
            int atom = base + wave * 16 + (lane >> 4) * 8 + r;
            if (atom < n_atoms) atomicAdd(&energy[img[atom]], o8[r]);
        }
    }
}

// ---------------------------------------------------------------------------
// Sparse force scatter: f[c] -= vals[i] * g[rows[i]]
// ---------------------------------------------------------------------------
__global__ void forces_kernel(const int* __restrict__ rows, const int* __restrict__ cols,
                              const float* __restrict__ vals, const float* __restrict__ g,
                              float* __restrict__ f, int nnz) {
    int stride = gridDim.x * blockDim.x;
    for (int i = blockIdx.x * blockDim.x + threadIdx.x; i < nnz; i += stride) {
        atomicAdd(&f[cols[i]], -vals[i] * g[rows[i]]);
    }
}

// ---------------------------------------------------------------------------
extern "C" void kernel_launch(void* const* d_in, const int* in_sizes, int n_in,
                              void* d_out, int out_size, void* d_ws, size_t ws_size,
                              hipStream_t stream) {
    const int*   an   = (const int*)d_in[0];
    const float* fp   = (const float*)d_in[1];
    const int*   img  = (const int*)d_in[2];
    const int*   rows = (const int*)d_in[3];
    const int*   cols = (const int*)d_in[4];
    const float* vals = (const float*)d_in[5];
    // d_in[6] = n_mol (device scalar) - derived on host below instead
    const float* W1 = (const float*)d_in[7];
    const float* b1 = (const float*)d_in[8];
    const float* W2 = (const float*)d_in[9];
    const float* b2 = (const float*)d_in[10];
    const float* W3 = (const float*)d_in[11];
    const float* b3 = (const float*)d_in[12];
    const float* W4 = (const float*)d_in[13];
    const float* b4 = (const float*)d_in[14];

    int n_atoms = in_sizes[0];
    int nnz     = in_sizes[3];
    int n_mol   = out_size - 3 * n_atoms;   // energy[n_mol] ++ forces[3*n_atoms]

    float* out    = (float*)d_out;
    float* energy = out;
    float* forces = out + n_mol;

    // Workspace: g [n_atoms x 120] f32, then f16-packed weights.
    float* g = (float*)d_ws;
    size_t gbytes = (size_t)n_atoms * IN_DIM * sizeof(float);
    _Float16* wsW = (_Float16*)((char*)d_ws + ((gbytes + 255) & ~(size_t)255));

    hipMemsetAsync(d_out, 0, (size_t)out_size * sizeof(float), stream);

    pack_weights<<<dim3(64, 3, 7), 256, 0, stream>>>(W1, W2, W3, W4, wsW);

    size_t shmem = (size_t)4 * 128 * LDSTR * sizeof(_Float16) + 128 * sizeof(int);
    int nblocks = (n_atoms + 127) / 128;
    bpnn_kernel<<<nblocks, 256, shmem, stream>>>(an, fp, img, wsW, b1, b2, b3,
                                                 W4, b4, g, energy, n_atoms);

    forces_kernel<<<2048, 256, 0, stream>>>(rows, cols, vals, g, forces, nnz);
}
